// MemoryController_10892037062777
// MI455X (gfx1250) — matrix operand order repaired and checked by reading, not compiled, not run on hardware
//
#include <hip/hip_runtime.h>

typedef __attribute__((ext_vector_type(16))) __bf16 v16bf;
typedef __attribute__((ext_vector_type(2)))  __bf16 v2bf;
typedef __attribute__((ext_vector_type(8)))  float  v8f;

// ---------------- WMMA fragment helpers (contiguous k = 16*(lane/16) + s, wave32) ----------

// A: 16x32 bf16 tile, row-major in LDS, leading dim `ld` (elements).
__device__ __forceinline__ v16bf frag_a(const __bf16* base, int ld) {
  int lane = threadIdx.x & 31;
  const __bf16* p = base + (lane & 15) * ld + ((lane >> 4) << 4);
  v16bf f;
#pragma unroll
  for (int i = 0; i < 8; ++i) f[i] = p[i];
#pragma unroll
  for (int i = 0; i < 8; ++i) f[i + 8] = p[8 + i];
  return f;
}

// B: 32x16 bf16 tile stored [N][K] row-major (contiguous 32B per lane).
__device__ __forceinline__ v16bf frag_b_nk(const __bf16* base, int ld) {
  int lane = threadIdx.x & 31;
  const __bf16* p = base + (lane & 15) * ld + ((lane >> 4) << 4);
  v16bf f;
#pragma unroll
  for (int i = 0; i < 16; ++i) f[i] = p[i];
  return f;
}

// B: 32x16 bf16 tile stored [K][N] row-major (strided per-lane gather).
__device__ __forceinline__ v16bf frag_b_kn(const __bf16* base, int ld) {
  int lane = threadIdx.x & 31;
  const __bf16* p = base + ((lane >> 4) << 4) * ld + (lane & 15);
  v16bf f;
#pragma unroll
  for (int i = 0; i < 16; ++i) f[i] = p[i * ld];
  return f;
}

__device__ __forceinline__ v8f wmma_bf16(v16bf a, v16bf b, v8f c) {
  return __builtin_amdgcn_wmma_f32_16x16x32_bf16(false, a, false, b, (short)0, c,
                                                 false, false);
}

// ---- CDNA5 async global->LDS copy (ASYNCcnt-tracked), 16 bytes per lane -----
__device__ __forceinline__ void async_copy_b128(void* lds_dst, const void* gsrc) {
  unsigned lds = (unsigned)(uintptr_t)lds_dst;  // addr[31:0] = LDS byte offset
  unsigned long long ga = (unsigned long long)(uintptr_t)gsrc;
  asm volatile("global_load_async_to_lds_b128 %0, %1, off"
               :: "v"(lds), "v"(ga) : "memory");
}
__device__ __forceinline__ void async_wait0() {
  asm volatile("s_wait_asynccnt 0x0" ::: "memory");
}

__device__ __forceinline__ float fast_rcp(float v) {
  return __builtin_amdgcn_rcpf(v);
}

// ---------------- Kernel 0: LayerNorm -> bf16 workspace -----------------------
// grid 256, block 256 (8 waves, 8 tokens per wave).
__global__ __launch_bounds__(256) void ln_convert_kernel(
    const float* __restrict__ x, const float* __restrict__ ln_w,
    const float* __restrict__ ln_b, __bf16* __restrict__ xln) {
  const int wave = threadIdx.x >> 5, lane = threadIdx.x & 31;
  for (int tl = 0; tl < 8; ++tl) {
    long t = (long)blockIdx.x * 64 + wave * 8 + tl;
    const float* xp = x + t * 2048;
    float2 v[32];
    float s = 0.f, s2 = 0.f;
#pragma unroll
    for (int u = 0; u < 32; ++u) {
      v[u] = ((const float2*)xp)[u * 32 + lane];
      s += v[u].x + v[u].y;
      s2 += v[u].x * v[u].x + v[u].y * v[u].y;
    }
#pragma unroll
    for (int m = 16; m; m >>= 1) { s += __shfl_xor(s, m); s2 += __shfl_xor(s2, m); }
    float mu = s * (1.f / 2048.f);
    float rs = rsqrtf(s2 * (1.f / 2048.f) - mu * mu + 1e-5f);
#pragma unroll
    for (int u = 0; u < 32; ++u) {
      int c = u * 64 + lane * 2;
      v2bf pk;
      pk[0] = (__bf16)((v[u].x - mu) * rs * ln_w[c] + ln_b[c]);
      pk[1] = (__bf16)((v[u].y - mu) * rs * ln_w[c + 1] + ln_b[c + 1]);
      *(v2bf*)(xln + t * 2048 + c) = pk;
    }
  }
}

// ---------------- Kernel 1: gate MLP partial logits ---------------------------
// grid 512 = 128 token-tiles x 4 N-splits; block 256. Single K pass,
// 16 accumulator fragments per wave (N=256). Partial logits combined with
// global atomics into a zeroed buffer.
__global__ __launch_bounds__(256) void gate_kernel(
    const __bf16* __restrict__ xln, const float* __restrict__ g1_w,
    const float* __restrict__ g1_b, const float* __restrict__ g2_w,
    float* __restrict__ logit_out) {
  __shared__ __bf16 As[128 * 32];    // LN(x) tile
  __shared__ __bf16 Bs[256 * 32];    // g1_w^T tile [256 N][32 K]
  __shared__ float logitS[128];
  const int tid = threadIdx.x, wave = tid >> 5, lane = tid & 31;
  const long tokBase = (long)(blockIdx.x >> 2) * 128;
  const int n0 = (blockIdx.x & 3) * 256;

  if (tid < 128) logitS[tid] = 0.f;

  v8f acc[16];
#pragma unroll
  for (int nt = 0; nt < 16; ++nt) acc[nt] = {};

  for (int k0 = 0; k0 < 2048; k0 += 32) {
    __syncthreads();
    // A tile: pure bf16 copy -> async DMA to LDS
#pragma unroll
    for (int ch = tid; ch < 512; ch += 256) {
      int r = ch >> 2, c = (ch & 3) << 3;
      async_copy_b128(As + r * 32 + c, xln + (tokBase + r) * 2048 + k0 + c);
    }
    // B tile: f32 -> bf16 transpose-stage, Bs[n][k]
    {
      const float* gp = g1_w + (long)k0 * 1024 + n0 + tid;
#pragma unroll
      for (int k = 0; k < 32; ++k) Bs[tid * 32 + k] = (__bf16)gp[(long)k * 1024];
    }
    async_wait0();
    __syncthreads();
    v16bf af = frag_a(As + (wave << 4) * 32, 32);
#pragma unroll
    for (int nt = 0; nt < 16; ++nt) {
      v16bf bf = frag_b_nk(Bs + (nt << 4) * 32, 32);
      acc[nt] = wmma_bf16(af, bf, acc[nt]);
    }
  }

  // SiLU + g2 dot; per-lane partials, then 16-lane shuffle reduce
  float part[8];
#pragma unroll
  for (int i = 0; i < 8; ++i) part[i] = 0.f;
#pragma unroll
  for (int nt = 0; nt < 16; ++nt) {
    int n = n0 + (nt << 4) + (lane & 15);
    float g1b = g1_b[n], g2v = g2_w[n];
#pragma unroll
    for (int i = 0; i < 8; ++i) {
      float h = acc[nt][i] + g1b;
      part[i] += h * fast_rcp(1.f + __expf(-h)) * g2v;  // silu(h) * g2
    }
  }
#pragma unroll
  for (int i = 0; i < 8; ++i) {
#pragma unroll
    for (int m = 8; m; m >>= 1) part[i] += __shfl_xor(part[i], m);
    if ((lane & 15) == 0) {
      int r = (wave << 4) + i + ((lane >> 4) << 3);
      atomicAdd(&logitS[r], part[i]);
    }
  }
  __syncthreads();
  if (tid < 128) atomicAdd(&logit_out[tokBase + tid], logitS[tid]);
}

// ---------------- Kernel 2: query = x @ rq_w + rq_b (bf16 out), one K pass ----
__global__ __launch_bounds__(256) void query_kernel(
    const float* __restrict__ x, const float* __restrict__ rq_w,
    const float* __restrict__ rq_b, __bf16* __restrict__ q_out) {
  __shared__ __bf16 As[128 * 32];
  __shared__ __bf16 Bs[256 * 32];
  const int tid = threadIdx.x, wave = tid >> 5, lane = tid & 31;
  const long tokBase = (long)blockIdx.x * 128;

  v8f acc[16];
#pragma unroll
  for (int nt = 0; nt < 16; ++nt) acc[nt] = {};

  for (int k0 = 0; k0 < 2048; k0 += 32) {
    __syncthreads();
    {
      int r = tid >> 1, kb = (tid & 1) << 4;
      const float* xp = x + (tokBase + r) * 2048 + k0 + kb;
#pragma unroll
      for (int j = 0; j < 16; ++j) As[r * 32 + kb + j] = (__bf16)xp[j];
    }
    {
      const float* gp = rq_w + (long)k0 * 256 + tid;
#pragma unroll
      for (int k = 0; k < 32; ++k) Bs[tid * 32 + k] = (__bf16)gp[(long)k * 256];
    }
    __syncthreads();
    v16bf af = frag_a(As + (wave << 4) * 32, 32);
#pragma unroll
    for (int nt = 0; nt < 16; ++nt) {
      v16bf bf = frag_b_nk(Bs + (nt << 4) * 32, 32);
      acc[nt] = wmma_bf16(af, bf, acc[nt]);
    }
  }
#pragma unroll
  for (int nt = 0; nt < 16; ++nt) {
    int n = (nt << 4) + (lane & 15);
    float b = rq_b[n];
#pragma unroll
    for (int i = 0; i < 8; ++i) {
      int r = (wave << 4) + i + ((lane >> 4) << 3);
      q_out[(tokBase + r) * 256 + n] = (__bf16)(acc[nt][i] + b);
    }
  }
}

// ---------------- Kernel 3: attention + readback + projection -----------------
// grid 256 (64 tokens/WG), block 256 (8 waves). Dynamic LDS ~142 KB.
static const int K3_SMEM = (288 * 128 + 64 * 288 + 64 * 128 + 32 * 288) * 2 + 64 * 4;

__global__ __launch_bounds__(256) void attend_kernel(
    const float* __restrict__ x, const float* __restrict__ cache,
    const __bf16* __restrict__ q_in, const float* __restrict__ logit_in,
    const float* __restrict__ g2_b, const float* __restrict__ fc_w,
    const float* __restrict__ fc_b, float* __restrict__ out_x,
    float* __restrict__ out_ctx, float* __restrict__ out_attn) {
  extern __shared__ char smemraw[];
  __bf16* localT = (__bf16*)smemraw;          // [288 d][128 slot] (d>=273 zero)
  __bf16* qc2    = localT + 288 * 128;        // [64][288]: query, then context_full
  __bf16* attnS  = qc2 + 64 * 288;            // [64][128] attn bf16
  __bf16* fcwT   = attnS + 64 * 128;          // [32 n][288 k] fc_w^T tile
  float*  gates  = (float*)(fcwT + 32 * 288); // [64]

  const int tid = threadIdx.x, wave = tid >> 5, lane = tid & 31;
  const int blk = blockIdx.x, b = blk >> 6;
  const long tokBase = (long)blk * 64;

  // Stage local cache slots, transposed: localT[d][slot]
  for (int idx = tid; idx < 288 * 128; idx += 256) {
    int slot = idx / 288, d = idx - slot * 288;
    float v = (d < 273) ? cache[((long)b * 1536 + 384 + slot) * 273 + d] : 0.f;
    localT[d * 128 + slot] = (__bf16)v;
  }
  // Stage query tile: pure bf16 copy -> async DMA to LDS (32 chunks per row)
#pragma unroll
  for (int ch = tid; ch < 64 * 32; ch += 256) {
    int r = ch >> 5, c = (ch & 31) << 3;
    async_copy_b128(qc2 + r * 288 + c, q_in + (tokBase + r) * 256 + c);
  }
  if (tid < 64) {
    float logit = logit_in[tokBase + tid] + g2_b[0];
    gates[tid] = (logit > 0.f) ? 1.f : 0.f;  // sigmoid(l) > 0.5  <=>  l > 0
  }
  async_wait0();
  __syncthreads();

  // Scores + softmax: waves 0..3 each own a 16-token M tile and all 128 slots
  if (wave < 4) {
    float sc[8][8];
    for (int nt = 0; nt < 8; ++nt) {
      v8f acc = {};
      for (int k0 = 0; k0 < 256; k0 += 32) {
        v16bf af = frag_a(qc2 + (wave << 4) * 288 + k0, 288);
        v16bf bf = frag_b_kn(localT + k0 * 128 + (nt << 4), 128);
        acc = wmma_bf16(af, bf, acc);
      }
#pragma unroll
      for (int i = 0; i < 8; ++i) sc[nt][i] = acc[i];
    }
#pragma unroll
    for (int i = 0; i < 8; ++i) {
      float m = -1e30f;
#pragma unroll
      for (int nt = 0; nt < 8; ++nt) {
        float v = fminf(20.f, fmaxf(-20.f, sc[nt][i] * 0.0625f));
        sc[nt][i] = v;
        m = fmaxf(m, v);
      }
#pragma unroll
      for (int msk = 8; msk; msk >>= 1) m = fmaxf(m, __shfl_xor(m, msk));
      float s = 0.f;
#pragma unroll
      for (int nt = 0; nt < 8; ++nt) {
        float e = __expf(sc[nt][i] - m);
        sc[nt][i] = e;
        s += e;
      }
#pragma unroll
      for (int msk = 8; msk; msk >>= 1) s += __shfl_xor(s, msk);
      float inv = fast_rcp(s);
      int r = (wave << 4) + i + ((lane >> 4) << 3);
#pragma unroll
      for (int nt = 0; nt < 8; ++nt) {
        float a = sc[nt][i] * inv;
        int col = (nt << 4) + (lane & 15);
        out_attn[(tokBase + r) * 128 + col] = a;
        attnS[r * 128 + col] = (__bf16)a;
      }
    }
  }
  __syncthreads();

  // GEMM2: context_full = attn(64x128) @ local(128x273) -> qc2 (bf16, 288-pad)
  {
    int mt = wave & 3, nh = wave >> 2;
    for (int nt = nh * 9; nt < nh * 9 + 9; ++nt) {
      v8f acc = {};
      for (int k0 = 0; k0 < 128; k0 += 32) {
        v16bf af = frag_a(attnS + (mt << 4) * 128 + k0, 128);
        v16bf bf = frag_b_nk(localT + (nt << 4) * 128 + k0, 128);
        acc = wmma_bf16(af, bf, acc);
      }
      int col = (nt << 4) + (lane & 15);
#pragma unroll
      for (int i = 0; i < 8; ++i) {
        int r = (mt << 4) + i + ((lane >> 4) << 3);
        qc2[r * 288 + col] = (__bf16)acc[i];
      }
    }
  }

  // GEMM3: context = context_full(64x288) @ fc_w(273x2048) + bias, fused epilogue
  const int mt = wave & 3, nh = wave >> 2;
  for (int n0 = 0; n0 < 2048; n0 += 32) {
    __syncthreads();  // also orders GEMM2 writes before first read
    for (int idx = tid; idx < 32 * 288; idx += 256) {
      int k = idx >> 5, n = idx & 31;
      float v = (k < 273) ? fc_w[(long)k * 2048 + n0 + n] : 0.f;
      fcwT[n * 288 + k] = (__bf16)v;
    }
    if (n0 + 32 < 2048 && tid < 256)  // prefetch next weight tile
      __builtin_prefetch(fc_w + (long)tid * 2048 + n0 + 32, 0, 0);
    __syncthreads();
    v8f acc = {};
    for (int k0 = 0; k0 < 288; k0 += 32) {
      v16bf af = frag_a(qc2 + (mt << 4) * 288 + k0, 288);
      v16bf bf = frag_b_nk(fcwT + (nh << 4) * 288 + k0, 288);
      acc = wmma_bf16(af, bf, acc);
    }
    int n = n0 + (nh << 4) + (lane & 15);
    float bias = fc_b[n];
#pragma unroll
    for (int i = 0; i < 8; ++i) {
      int r = (mt << 4) + i + ((lane >> 4) << 3);
      long t = tokBase + r;
      float ctx = acc[i] + bias;
      float xv = x[t * 2048 + n];
      out_ctx[t * 2048 + n] = ctx;
      out_x[t * 2048 + n] = xv + gates[r] * ctx;
    }
  }
}

// -----------------------------------------------------------------------------
extern "C" void kernel_launch(void* const* d_in, const int* in_sizes, int n_in,
                              void* d_out, int out_size, void* d_ws, size_t ws_size,
                              hipStream_t stream) {
  (void)in_sizes; (void)n_in; (void)out_size; (void)ws_size;
  const float* x     = (const float*)d_in[0];
  const float* cache = (const float*)d_in[1];
  const float* ln_w  = (const float*)d_in[2];
  const float* ln_b  = (const float*)d_in[3];
  const float* g1_w  = (const float*)d_in[4];
  const float* g1_b  = (const float*)d_in[5];
  const float* g2_w  = (const float*)d_in[6];
  const float* g2_b  = (const float*)d_in[7];
  const float* rq_w  = (const float*)d_in[8];
  const float* rq_b  = (const float*)d_in[9];
  const float* fc_w  = (const float*)d_in[10];
  const float* fc_b  = (const float*)d_in[11];

  float* out_x   = (float*)d_out;
  float* out_ctx = out_x + (size_t)4 * 4096 * 2048;
  float* out_at  = out_ctx + (size_t)4 * 4096 * 2048;

  // workspace: [logits f32 64KB][query bf16 8MB][xln bf16 64MB]
  float*  logit_ws = (float*)d_ws;
  __bf16* q_ws     = (__bf16*)((char*)d_ws + 16384 * 4);
  __bf16* xln_ws   = (__bf16*)((char*)d_ws + 16384 * 4 + (size_t)16384 * 256 * 2);

  hipMemsetAsync(logit_ws, 0, 16384 * sizeof(float), stream);
  ln_convert_kernel<<<256, 256, 0, stream>>>(x, ln_w, ln_b, xln_ws);
  gate_kernel<<<512, 256, 0, stream>>>(xln_ws, g1_w, g1_b, g2_w, logit_ws);
  query_kernel<<<128, 256, 0, stream>>>(x, rq_w, rq_b, q_ws);
  attend_kernel<<<256, 256, K3_SMEM, stream>>>(x, cache, q_ws, logit_ws, g2_b,
                                               fc_w, fc_b, out_x, out_ctx, out_at);
}
